// Net_69252052680906
// MI455X (gfx1250) — matrix operand order repaired and checked by reading, not compiled, not run on hardware
//
#include <hip/hip_runtime.h>
#include <hip/hip_bf16.h>
#include <math.h>

// ---------------------------------------------------------------------------
// CDNA5 (gfx1250) wave32 WMMA types
// ---------------------------------------------------------------------------
typedef __attribute__((ext_vector_type(16))) __bf16 v16bf;
typedef __attribute__((ext_vector_type(8)))  float  v8f;

#define LEAKY_SLOPE 0.2f
#define BN_EPS      1e-5f
// order-preserving float->uint encoding for atomic max on floats
#define ENC_NEG_INF 0x007FFFFFu   // fenc(-inf)

static __device__ __forceinline__ unsigned fenc(float f) {
  unsigned u = __float_as_uint(f);
  return (u & 0x80000000u) ? ~u : (u | 0x80000000u);
}
static __device__ __forceinline__ float fdec(unsigned e) {
  unsigned u = (e & 0x80000000u) ? (e & 0x7FFFFFFFu) : ~e;
  return __uint_as_float(u);
}

// ---------------------------------------------------------------------------
// fill
// ---------------------------------------------------------------------------
__global__ void k_fill_u32(unsigned* __restrict__ p, unsigned v, long long n) {
  long long i = blockIdx.x * (long long)blockDim.x + threadIdx.x;
  if (i < n) p[i] = v;
}

// ---------------------------------------------------------------------------
// out[M,Nc] = A[M,K] @ W[Nc,K]^T    fp32 in/out, bf16 WMMA, f32 accumulate.
// grid = (ceil(M/16), Nc/64), block = 128 (4 waves; wave w owns 16 columns).
// K in {64,128}; Nc multiple of 64. Rows >= M are clamped on load and
// guarded on store (fc5 path, M=100).
// ---------------------------------------------------------------------------
__global__ void k_gemm_bf16_wmma(const float* __restrict__ A,
                                 const float* __restrict__ W,
                                 float* __restrict__ out,
                                 int M, int K, int Nc) {
  __shared__ __bf16 As[16 * 128];   // A tile  (16 rows x K)
  __shared__ __bf16 Ws[64 * 128];   // W tile  (64 out-cols x K)
  const int tid = threadIdx.x;
  const int m0  = blockIdx.x * 16;
  const int j0  = blockIdx.y * 64;

  for (int i = tid; i < 16 * K; i += 128) {
    int r = i / K, k = i - r * K;
    int gr = m0 + r; if (gr >= M) gr = M - 1;      // clamp (store is guarded)
    As[i] = (__bf16)A[(long long)gr * K + k];
  }
  for (int i = tid; i < 64 * K; i += 128) {
    int j = i / K, k = i - j * K;
    Ws[i] = (__bf16)W[(long long)(j0 + j) * K + k];
  }
  __syncthreads();

  const int lane   = tid & 31;
  const int wv     = tid >> 5;       // 0..3
  const int laneHi = lane >> 4;      // 0/1
  const int lm     = lane & 15;

  v8f c = {};
  for (int kb = 0; kb < K; kb += 32) {
    v16bf a, b;
#pragma unroll
    for (int e = 0; e < 16; ++e) {
      // A (16x32, MxK): lanes 0-15 -> K {0..7,16..23}; lanes 16-31 -> +8
      int ka = kb + ((e >> 3) << 4) + (e & 7) + (laneHi << 3);
      a[e] = As[lm * K + ka];
      // B (32x16, KxN): lane = column; lanes 0-15 -> K {0..7,16..23}; lanes 16-31 -> +8
      int kbv = kb + ((e >> 3) << 4) + (e & 7) + (laneHi << 3);
      b[e] = Ws[(wv * 16 + lm) * K + kbv];
    }
    c = __builtin_amdgcn_wmma_f32_16x16x32_bf16(false, a, false, b,
                                                (short)0, c, false, false);
  }

  const int col = j0 + wv * 16 + lm;
#pragma unroll
  for (int r = 0; r < 8; ++r) {
    int row = m0 + r + laneHi * 8;   // D: VGPR r -> M=r (lanes 0-15) / r+8
    if (row < M) out[(long long)row * Nc + col] = c[r];
  }
}

// ---------------------------------------------------------------------------
// a_s[n,h] = sum_c h[n,h,c]*att_src[h,c] ; a_d likewise (H = 2)
// ---------------------------------------------------------------------------
__global__ void k_attn_coef(const float* __restrict__ h,
                            const float* __restrict__ asrc,
                            const float* __restrict__ adst,
                            float* __restrict__ as_, float* __restrict__ ad_,
                            int N, int C) {
  int n = blockIdx.x * blockDim.x + threadIdx.x;
  if (n >= N) return;
  const float* row = h + (long long)n * 2 * C;
  for (int hh = 0; hh < 2; ++hh) {
    float s = 0.f, d = 0.f;
    for (int cc = 0; cc < C; ++cc) {
      float v = row[hh * C + cc];
      s += v * asrc[hh * C + cc];
      d += v * adst[hh * C + cc];
    }
    as_[n * 2 + hh] = s;
    ad_[n * 2 + hh] = d;
  }
}

// ---------------------------------------------------------------------------
// edge range is [0,E) real edges, [E,E+N) self loops
// ---------------------------------------------------------------------------
__global__ void k_edge_max(const int* __restrict__ src, const int* __restrict__ dst,
                           const float* __restrict__ as_, const float* __restrict__ ad_,
                           unsigned* __restrict__ menc, int E, int N) {
  int i = blockIdx.x * blockDim.x + threadIdx.x;
  if (i >= E + N) return;
  int s, d;
  if (i < E) { s = src[i]; d = dst[i]; } else { s = d = i - E; }
#pragma unroll
  for (int hh = 0; hh < 2; ++hh) {
    float v = as_[s * 2 + hh] + ad_[d * 2 + hh];
    v = v > 0.f ? v : LEAKY_SLOPE * v;
    atomicMax(&menc[d * 2 + hh], fenc(v));
  }
}

__global__ void k_edge_expsum(const int* __restrict__ src, const int* __restrict__ dst,
                              const float* __restrict__ as_, const float* __restrict__ ad_,
                              const unsigned* __restrict__ menc,
                              float* __restrict__ ex, float* __restrict__ den,
                              int E, int N) {
  int i = blockIdx.x * blockDim.x + threadIdx.x;
  if (i >= E + N) return;
  int s, d;
  if (i < E) { s = src[i]; d = dst[i]; } else { s = d = i - E; }
#pragma unroll
  for (int hh = 0; hh < 2; ++hh) {
    float v = as_[s * 2 + hh] + ad_[d * 2 + hh];
    v = v > 0.f ? v : LEAKY_SLOPE * v;
    float t = expf(v - fdec(menc[d * 2 + hh]));
    ex[(long long)i * 2 + hh] = t;
    atomicAdd(&den[d * 2 + hh], t);
  }
}

// agg[d, c] += (ex/den) * h[s, c];  thread = (edge, channel) so a 128-thread
// group streams one contiguous feature row (coalesced 512B gathers).
__global__ void k_edge_agg(const int* __restrict__ src, const int* __restrict__ dst,
                           const float* __restrict__ hfeat,
                           const float* __restrict__ ex, const float* __restrict__ den,
                           float* __restrict__ agg, int E, int N, int C) {
  long long idx = blockIdx.x * (long long)blockDim.x + threadIdx.x;
  const int HC = 2 * C;
  long long total = (long long)(E + N) * HC;
  if (idx >= total) return;
  int e = (int)(idx / HC);
  int c = (int)(idx - (long long)e * HC);
  int hh = (c >= C) ? 1 : 0;
  int s, d;
  if (e < E) { s = src[e]; d = dst[e]; } else { s = d = e - E; }
  float alpha = ex[(long long)e * 2 + hh] / den[d * 2 + hh];
  atomicAdd(&agg[(long long)d * HC + c], alpha * hfeat[(long long)s * HC + c]);
}

// ---------------------------------------------------------------------------
// BatchNorm (training-mode biased stats) over y = relu(in + bias)
// ---------------------------------------------------------------------------
__global__ void k_bn_reduce(const float* __restrict__ in, const float* __restrict__ bias,
                            float* __restrict__ stats, int N, int C, int rowsPerBlock) {
  __shared__ float ssum[256], ssq[256];
  int t = threadIdx.x;
  int c = t % C;
  int rstep = 256 / C;                         // C in {64,128,256}
  int rEnd = (blockIdx.x + 1) * rowsPerBlock; if (rEnd > N) rEnd = N;
  float a = 0.f, b = 0.f;
  for (int r = blockIdx.x * rowsPerBlock + t / C; r < rEnd; r += rstep) {
    float y = in[(long long)r * C + c] + bias[c];
    y = y > 0.f ? y : 0.f;
    a += y; b += y * y;
  }
  ssum[t] = a; ssq[t] = b;
  __syncthreads();
  if (t < C) {
    for (int u = t + C; u < 256; u += C) { a += ssum[u]; b += ssq[u]; }
    atomicAdd(&stats[c], a);
    atomicAdd(&stats[C + c], b);
  }
}

__global__ void k_bn_apply(const float* __restrict__ in, const float* __restrict__ bias,
                           const float* __restrict__ stats,
                           const float* __restrict__ gamma, const float* __restrict__ beta,
                           float* __restrict__ out, int N, int C) {
  long long i = blockIdx.x * (long long)blockDim.x + threadIdx.x;
  if (i >= (long long)N * C) return;
  int c = (int)(i % C);
  float y = in[i] + bias[c];
  y = y > 0.f ? y : 0.f;
  float mu  = stats[c] / (float)N;
  float var = stats[C + c] / (float)N - mu * mu;
  out[i] = (y - mu) * rsqrtf(var + BN_EPS) * gamma[c] + beta[c];
}

// ---------------------------------------------------------------------------
// sort pool: stable argsort of -last_channel per graph via rank counting
// ---------------------------------------------------------------------------
__global__ void k_sort_pool(const float* __restrict__ post, float* __restrict__ sel,
                            int NPG, int C) {
  __shared__ float v[1000];
  int g = blockIdx.x, t = threadIdx.x;
  const float* base = post + (long long)g * NPG * C;
  for (int i = t; i < NPG; i += blockDim.x) v[i] = base[(long long)i * C + (C - 1)];
  __syncthreads();
  for (int i = t; i < NPG; i += blockDim.x) {
    float vi = v[i];
    int rank = 0;
    for (int j = 0; j < NPG; ++j) {
      float vj = v[j];
      rank += (vj > vi) || (vj == vi && j < i);
    }
    sel[(long long)g * NPG + rank] = (float)i;
  }
}

__global__ void k_mean_pool(const float* __restrict__ post, float* __restrict__ pooled,
                            int NPG, int C) {
  int g = blockIdx.x, c = threadIdx.x;   // blockDim = C (64)
  const float* base = post + (long long)g * NPG * C;
  float s = 0.f;
  for (int r = 0; r < NPG; ++r) s += base[(long long)r * C + c];
  pooled[g * C + c] = s / (float)NPG;
}

// logits = zbn @ wout^T + bout ; log_softmax over the 2 classes
__global__ void k_head(const float* __restrict__ zbn, const float* __restrict__ wout,
                       const float* __restrict__ bout, float* __restrict__ out,
                       int B, int K) {
  int b = blockIdx.x * blockDim.x + threadIdx.x;
  if (b >= B) return;
  float l0 = bout[0], l1 = bout[1];
  for (int k = 0; k < K; ++k) {
    float z = zbn[(long long)b * K + k];
    l0 += z * wout[k];
    l1 += z * wout[K + k];
  }
  float m = fmaxf(l0, l1);
  float lse = m + logf(expf(l0 - m) + expf(l1 - m));
  out[b * 2 + 0] = l0 - lse;
  out[b * 2 + 1] = l1 - lse;
}

// ---------------------------------------------------------------------------
// launch
// ---------------------------------------------------------------------------
extern "C" void kernel_launch(void* const* d_in, const int* in_sizes, int n_in,
                              void* d_out, int out_size, void* d_ws, size_t ws_size,
                              hipStream_t stream) {
  (void)n_in; (void)out_size; (void)ws_size;
  const float* x    = (const float*)d_in[0];
  const int*   ei   = (const int*)d_in[1];      // [2, E]
  const float* w1   = (const float*)d_in[3];
  const float* as1  = (const float*)d_in[4];
  const float* ad1  = (const float*)d_in[5];
  const float* b1   = (const float*)d_in[6];
  const float* g1   = (const float*)d_in[7];
  const float* be1  = (const float*)d_in[8];
  const float* w2   = (const float*)d_in[9];
  const float* as2  = (const float*)d_in[10];
  const float* ad2  = (const float*)d_in[11];
  const float* b2   = (const float*)d_in[12];
  const float* g2   = (const float*)d_in[13];
  const float* be2  = (const float*)d_in[14];
  const float* w5   = (const float*)d_in[15];
  const float* b5   = (const float*)d_in[16];
  const float* g256 = (const float*)d_in[17];
  const float* be256= (const float*)d_in[18];
  const float* wout = (const float*)d_in[19];
  const float* bout = (const float*)d_in[20];

  const int N   = in_sizes[0] / 64;
  const int E   = in_sizes[1] / 2;
  const int B   = 100;
  const int NPG = N / B;
  const int EP  = E + N;
  const int* esrc = ei;
  const int* edst = ei + E;

  // workspace layout
  char* ws = (char*)d_ws;
  size_t o = 0;
  auto take = [&](size_t bytes) { char* p = ws + o; o = (o + bytes + 255) & ~(size_t)255; return p; };
  float*    h_pre  = (float*)take((size_t)N * 128 * 4);
  float*    agg    = (float*)take((size_t)N * 128 * 4);
  float*    post   = (float*)take((size_t)N * 128 * 4);
  float*    as_    = (float*)take((size_t)N * 2 * 4);
  float*    ad_    = (float*)take((size_t)N * 2 * 4);
  unsigned* menc   = (unsigned*)take((size_t)N * 2 * 4);
  float*    den    = (float*)take((size_t)N * 2 * 4);
  float*    exbuf  = (float*)take((size_t)EP * 2 * 4);
  float*    stats  = (float*)take(512 * 4);
  float*    pooled = (float*)take((size_t)B * 64 * 4);
  float*    zfc    = (float*)take((size_t)B * 256 * 4);
  float*    zbn    = (float*)take((size_t)B * 256 * 4);

  auto blocks = [](long long n, int bs) { return (unsigned)((n + bs - 1) / bs); };

  auto run_conv = [&](const float* feat, const float* W, int K, int C,
                      const float* asrc, const float* adst, const float* bias,
                      const float* gamma, const float* beta, float* post_out) {
    const int HC = 2 * C;
    dim3 gg(blocks(N, 16), HC / 64);
    k_gemm_bf16_wmma<<<gg, 128, 0, stream>>>(feat, W, h_pre, N, K, HC);
    k_attn_coef<<<blocks(N, 256), 256, 0, stream>>>(h_pre, asrc, adst, as_, ad_, N, C);
    k_fill_u32<<<blocks((long long)N * 2, 256), 256, 0, stream>>>(menc, ENC_NEG_INF, (long long)N * 2);
    k_fill_u32<<<blocks((long long)N * 2, 256), 256, 0, stream>>>((unsigned*)den, 0u, (long long)N * 2);
    k_fill_u32<<<blocks((long long)N * HC, 256), 256, 0, stream>>>((unsigned*)agg, 0u, (long long)N * HC);
    k_edge_max<<<blocks(EP, 256), 256, 0, stream>>>(esrc, edst, as_, ad_, menc, E, N);
    k_edge_expsum<<<blocks(EP, 256), 256, 0, stream>>>(esrc, edst, as_, ad_, menc, exbuf, den, E, N);
    long long tot = (long long)EP * HC;
    k_edge_agg<<<blocks(tot, 256), 256, 0, stream>>>(esrc, edst, h_pre, exbuf, den, agg, E, N, C);
    k_fill_u32<<<1, 256, 0, stream>>>((unsigned*)stats, 0u, 2 * HC);
    const int rpb = 512;
    k_bn_reduce<<<blocks(N, rpb), 256, 0, stream>>>(agg, bias, stats, N, HC, rpb);
    k_bn_apply<<<blocks((long long)N * HC, 256), 256, 0, stream>>>(agg, bias, stats, gamma, beta, post_out, N, HC);
  };

  // conv1: GATConv(64 -> 64, H=2) + relu + BN
  run_conv(x, w1, 64, 64, as1, ad1, b1, g1, be1, post);
  // conv2: GATConv(128 -> 32, H=2) + relu + BN
  run_conv(post, w2, 128, 32, as2, ad2, b2, g2, be2, post);

  float* out_f = (float*)d_out;
  // select_index -> d_out[B*2 ..)
  k_sort_pool<<<B, 256, 0, stream>>>(post, out_f + (size_t)B * 2, NPG, 64);
  // mean pool -> pooled[B,64]
  k_mean_pool<<<B, 64, 0, stream>>>(post, pooled, NPG, 64);
  // fc5: z = pooled @ w5^T  (WMMA, M=100 padded tiles)
  k_gemm_bf16_wmma<<<dim3(blocks(B, 16), 256 / 64), 128, 0, stream>>>(pooled, w5, zfc, B, 64, 256);
  // BN(relu(z + b5))
  k_fill_u32<<<1, 512, 0, stream>>>((unsigned*)stats, 0u, 512);
  k_bn_reduce<<<1, 256, 0, stream>>>(zfc, b5, stats, B, 256, 512);
  k_bn_apply<<<blocks((long long)B * 256, 256), 256, 0, stream>>>(zfc, b5, stats, g256, be256, zbn, B, 256);
  // final linear + log_softmax -> d_out[0 .. B*2)
  k_head<<<1, 128, 0, stream>>>(zbn, wout, bout, out_f, B, 256);
}
